// RecurrentLayer_7808250544914
// MI455X (gfx1250) — matrix operand order repaired and checked by reading, not compiled, not run on hardware
//
#include <hip/hip_runtime.h>
#include <hip/hip_bf16.h>

// ---- CDNA5 (gfx1250, wave32) tanh-RNN scan using v_wmma_f32_16x16x32_bf16 ----
// D = Wi2h[128x256] x combined[256x16]  (8 M-tiles x 8 K-blocks of 32)
// out = Wi2o[80(pad)x128] x h[128x16]   (5 M-tiles x 4 K-blocks)
// One wave owns a 16-wide batch tile and scans all 512 timesteps.

typedef __attribute__((ext_vector_type(16))) __bf16 v16bf;
typedef __attribute__((ext_vector_type(8)))  float  v8f;

#define SEQ   512
#define BATCH 2048
#define IN    128
#define HID   128
#define NOUT  67
#define KCOMB 256

// dynamic LDS layout (bytes)
#define WA_ELEMS (8*8*32*16)            // Wi2h fragments, bf16
#define WO_ELEMS (5*4*32*16)            // Wi2o fragments, bf16 (rows >= 67 zero)
#define BH_ELEMS (8*32*8)               // hidden-bias C fragments, f32
#define BO_ELEMS (5*32*8)               // output-bias C fragments, f32
#define WA_OFF 0
#define WO_OFF (WA_ELEMS*2)             // 65536
#define BH_OFF (WO_OFF + WO_ELEMS*2)    // 86016
#define BO_OFF (BH_OFF + BH_ELEMS*4)    // 94208
#define SMEM_BYTES (BO_OFF + BO_ELEMS*4) // 99328  (<320KB/WG on CDNA5)

__global__ void __launch_bounds__(64)
rnn_wmma_kernel(const float* __restrict__ x,
                const float* __restrict__ hidden,
                const float* __restrict__ dropm,
                const float* __restrict__ Wi2h,
                const float* __restrict__ bi2h,
                const float* __restrict__ Wi2o,
                const float* __restrict__ bi2o,
                float* __restrict__ Y) {
  extern __shared__ char smem[];
  __bf16* wa = (__bf16*)(smem + WA_OFF);
  __bf16* wo = (__bf16*)(smem + WO_OFF);
  float*  bh = (float*) (smem + BH_OFF);
  float*  bo = (float*) (smem + BO_OFF);

  const int tid = threadIdx.x;
  const int nth = blockDim.x;

  // ---- stage weights/biases as exact WMMA fragments in LDS (once) ----
  // A fragments use contiguous k order here: lane<16 holds K=0..15, lane>=16 holds K=16..31
  for (int f = tid; f < WA_ELEMS; f += nth) {
    int e = f & 15, lane = (f >> 4) & 31, j = (f >> 9) & 7, t8 = f >> 12;
    int hi = (lane >> 4) << 3;
    int k  = 32*j + 2*hi + e;
    int m  = 16*t8 + (lane & 15);
    wa[f] = (__bf16)Wi2h[m*KCOMB + k];
  }
  for (int f = tid; f < WO_ELEMS; f += nth) {
    int e = f & 15, lane = (f >> 4) & 31, j = (f >> 9) & 3, t5 = f >> 11;
    int hi = (lane >> 4) << 3;
    int k  = 32*j + 2*hi + e;
    int m  = 16*t5 + (lane & 15);
    wo[f] = (m < NOUT) ? (__bf16)Wi2o[m*HID + k] : (__bf16)0.0f;
  }
  // C/D layout: lane<16 holds rows {0..7}, lane>=16 holds rows {8..15} of each M-tile
  for (int f = tid; f < BH_ELEMS; f += nth) {
    int r = f & 7, lane = (f >> 3) & 31, t8 = f >> 8;
    bh[f] = bi2h[16*t8 + ((lane >> 4) << 3) + r];
  }
  for (int f = tid; f < BO_ELEMS; f += nth) {
    int r = f & 7, lane = (f >> 3) & 31, t5 = f >> 8;
    int m = 16*t5 + ((lane >> 4) << 3) + r;
    bo[f] = (m < NOUT) ? bi2o[m] : 0.0f;
  }
  __syncthreads();

  const int lane = tid & 31;
  const int wave = tid >> 5;
  const int tile = blockIdx.x * (blockDim.x >> 5) + wave;  // 0..127
  const int col  = tile * 16 + (lane & 15);                // batch column
  const int hi2  = lane & 16;                              // B-fragment K offset (0/16)

  // initial recurrent state as B fragments: same contiguous k order as A, lane<16 -> K=0..15, lane>=16 -> K=16..31
  v16bf hB[4];
#pragma unroll
  for (int j = 0; j < 4; ++j) {
    v16bf hb;
#pragma unroll
    for (int e = 0; e < 16; ++e) {
      int hr = 32*j + hi2 + e;
      float v = hidden[hr*BATCH + col] * dropm[hr*BATCH + col];
      hb[e] = (__bf16)v;
    }
    hB[j] = hb;
  }

  for (int ts = 0; ts < SEQ; ++ts) {
    // ---- x_t B fragments (strided fp32 gather, 64B coalesced per half-wave) ----
    v16bf xB[4];
#pragma unroll
    for (int j = 0; j < 4; ++j) {
      v16bf xb;
#pragma unroll
      for (int e = 0; e < 16; ++e) {
        size_t k = (size_t)(32*j + hi2 + e);
        xb[e] = (__bf16)x[k*(size_t)(SEQ*BATCH) + (size_t)ts*BATCH + col];
      }
      xB[j] = xb;
    }

    // ---- h_new = tanh(Wi2h @ [x;h] + b): 8 M-tiles x 8 K-blocks ----
    v8f acc[8];
#pragma unroll
    for (int t8 = 0; t8 < 8; ++t8) {
      v8f a = *(const v8f*)&bh[(t8*32 + lane)*8];
#pragma unroll
      for (int j = 0; j < 4; ++j) {           // x part (K = 0..127)
        v16bf w = *(const v16bf*)&wa[((t8*8 + j)*32 + lane)*16];
        a = __builtin_amdgcn_wmma_f32_16x16x32_bf16(false, w, false, xB[j],
                                                    (short)0, a, false, false);
      }
#pragma unroll
      for (int j = 0; j < 4; ++j) {           // h part (K = 128..255)
        v16bf w = *(const v16bf*)&wa[((t8*8 + j + 4)*32 + lane)*16];
        a = __builtin_amdgcn_wmma_f32_16x16x32_bf16(false, w, false, hB[j],
                                                    (short)0, a, false, false);
      }
#pragma unroll
      for (int r = 0; r < 8; ++r) a[r] = tanhf(a[r]);
      acc[t8] = a;
    }

    // ---- rebuild B fragments from D fragments: one xor-16 lane swap ----
    // lane<16 e<8: D[2j][e]         e>=8: partner D[2j][e-8]
    // lane>=16 e<8: partner D[2j+1][e]   e>=8: D[2j+1][e-8]
#pragma unroll
    for (int j = 0; j < 4; ++j) {
      v16bf nb;
#pragma unroll
      for (int r = 0; r < 8; ++r) {
        float a_self = acc[2*j][r];
        float b_self = acc[2*j + 1][r];
        float a_sw = __shfl_xor(a_self, 16, 32);
        float b_sw = __shfl_xor(b_self, 16, 32);
        nb[r]     = (__bf16)((lane < 16) ? a_self : b_sw);
        nb[r + 8] = (__bf16)((lane < 16) ? a_sw   : b_self);
      }
      hB[j] = nb;
    }

    // ---- out = relu(Wi2o @ h + b): 5 M-tiles (pad 67->80) x 4 K-blocks ----
    const int hiD = (lane & 16) >> 1;
#pragma unroll
    for (int t5 = 0; t5 < 5; ++t5) {
      v8f oa = *(const v8f*)&bo[(t5*32 + lane)*8];
#pragma unroll
      for (int j = 0; j < 4; ++j) {
        v16bf w = *(const v16bf*)&wo[((t5*4 + j)*32 + lane)*16];
        oa = __builtin_amdgcn_wmma_f32_16x16x32_bf16(false, w, false, hB[j],
                                                     (short)0, oa, false, false);
      }
#pragma unroll
      for (int r = 0; r < 8; ++r) {
        int o = 16*t5 + hiD + r;
        if (o < NOUT) {
          float v = oa[r];
          Y[((size_t)o*SEQ + ts)*BATCH + col] = (v > 0.0f) ? v : 0.0f;
        }
      }
    }
  }
}

extern "C" void kernel_launch(void* const* d_in, const int* in_sizes, int n_in,
                              void* d_out, int out_size, void* d_ws, size_t ws_size,
                              hipStream_t stream) {
  (void)in_sizes; (void)n_in; (void)out_size; (void)d_ws; (void)ws_size;
  const float* x     = (const float*)d_in[0];
  const float* hid   = (const float*)d_in[1];
  const float* dropm = (const float*)d_in[2];
  const float* Wi2h  = (const float*)d_in[3];
  const float* bi2h  = (const float*)d_in[4];
  const float* Wi2o  = (const float*)d_in[5];
  const float* bi2o  = (const float*)d_in[6];
  float* Y = (float*)d_out;

  // 64 blocks x 64 threads (2 waves) -> 128 waves, one per 16-wide batch tile.
  dim3 grid(64), block(64);
  rnn_wmma_kernel<<<grid, block, SMEM_BYTES, stream>>>(
      x, hid, dropm, Wi2h, bi2h, Wi2o, bi2o, Y);
}
